// FlaxMptAttention_69758858822240
// MI455X (gfx1250) — matrix operand order repaired and checked by reading, not compiled, not run on hardware
//
#include <hip/hip_runtime.h>

// ---------------- problem constants ----------------
#define BB  2
#define SS  2048
#define DD  2048
#define HH  16
#define DHD 128            // head dim
#define F3D (3*DD)         // 6144
#define MROWS (BB*SS)      // 4096

typedef __attribute__((ext_vector_type(16))) __bf16 bf16x16;
typedef __attribute__((ext_vector_type(8)))  float  f32x8;
typedef __attribute__((ext_vector_type(4)))  unsigned u32x4;

union Frag {
    bf16x16 v;
    u32x4   u[2];
};

__device__ __forceinline__ unsigned short f2bf(float f) {
    union { float f; unsigned u; } c; c.f = f;
    unsigned u = c.u;
    unsigned r = u + 0x7FFFu + ((u >> 16) & 1u);   // round-to-nearest-even
    return (unsigned short)(r >> 16);
}

__device__ __forceinline__ void load_frag(Frag& f, const unsigned short* p0,
                                          const unsigned short* p1) {
    f.u[0] = *(const u32x4*)p0;
    f.u[1] = *(const u32x4*)p1;
}

// ---------------- elementwise f32 -> bf16 ----------------
__global__ __launch_bounds__(256) void cvt_bf16_kernel(const float* __restrict__ in,
                                                       unsigned short* __restrict__ out, int n) {
    for (int i = blockIdx.x * blockDim.x + threadIdx.x; i < n; i += gridDim.x * blockDim.x)
        out[i] = f2bf(in[i]);
}

// ---------------- tiled transpose + convert: out[c*rows + r] = bf16(in[r*cols + c]) ----------------
__global__ __launch_bounds__(256) void transpose_cvt_kernel(const float* __restrict__ in,
                                                            unsigned short* __restrict__ out,
                                                            int rows, int cols) {
    __shared__ float tile[32][33];
    const int c0 = blockIdx.x * 32, r0 = blockIdx.y * 32;
    const int tx = threadIdx.x & 31, ty = threadIdx.x >> 5;
    #pragma unroll
    for (int i = 0; i < 32; i += 8)
        tile[ty + i][tx] = in[(size_t)(r0 + ty + i) * cols + c0 + tx];
    __syncthreads();
    #pragma unroll
    for (int i = 0; i < 32; i += 8)
        out[(size_t)(c0 + ty + i) * rows + r0 + tx] = f2bf(tile[tx][ty + i]);
}

// ---------------- WMMA GEMM: C[M,N] = A[M,K](bf16,row) * Bt[N,K](bf16,row-of-cols) + bias ----------------
// block = 256 threads (8 waves). Block tile: 128(M) x 64(N). Wave tile: 16(M) x 64(N).
// Software-pipelined: next K-step fragments are loaded while the current WMMAs issue,
// plus global_prefetch one further tile ahead.
__global__ __launch_bounds__(256) void gemm_bf16_kernel(const unsigned short* __restrict__ A,
                                                        const unsigned short* __restrict__ Bt,
                                                        const float* __restrict__ bias,
                                                        float* __restrict__ C,
                                                        int M, int N, int K) {
    const int lane  = threadIdx.x & 31;
    const int wave  = threadIdx.x >> 5;
    const int m0    = blockIdx.y * 128 + wave * 16;
    const int n0    = blockIdx.x * 64;
    const int l15   = lane & 15;
    const int hi    = lane >> 4;           // 0: lanes 0-15, 1: lanes 16-31
    const int kbA   = hi ? 8 : 0;          // A-frag K phase
    const int ksB   = hi ? 8 : 0;          // B-frag K phase

    f32x8 acc[4] = {};
    const unsigned short* arow = A + (size_t)(m0 + l15) * K + kbA;
    const unsigned short* brow[4];
    #pragma unroll
    for (int t = 0; t < 4; ++t)
        brow[t] = Bt + (size_t)(n0 + t * 16 + l15) * K + ksB;

    // prologue: load K-step 0
    Frag a, b[4];
    load_frag(a, arow, arow + 16);
    #pragma unroll
    for (int t = 0; t < 4; ++t)
        load_frag(b[t], brow[t], brow[t] + 16);

    for (int k0 = 32; k0 < K; k0 += 32) {
        // stage next K-step fragments
        Frag an, bn[4];
        load_frag(an, arow + k0, arow + k0 + 16);
        #pragma unroll
        for (int t = 0; t < 4; ++t)
            load_frag(bn[t], brow[t] + k0, brow[t] + k0 + 16);
        // prefetch one tile further ahead (global_prefetch_b8)
        if (k0 + 32 < K) {
            __builtin_prefetch(arow + k0 + 32, 0, 1);
            #pragma unroll
            for (int t = 0; t < 4; ++t)
                __builtin_prefetch(brow[t] + k0 + 32, 0, 1);
        }
        // compute with current fragments (loads above are still in flight)
        #pragma unroll
        for (int t = 0; t < 4; ++t)
            acc[t] = __builtin_amdgcn_wmma_f32_16x16x32_bf16(false, a.v, false, b[t].v,
                                                             (short)0, acc[t], false, false);
        a = an;
        #pragma unroll
        for (int t = 0; t < 4; ++t) b[t] = bn[t];
    }
    // epilogue K-step
    #pragma unroll
    for (int t = 0; t < 4; ++t)
        acc[t] = __builtin_amdgcn_wmma_f32_16x16x32_bf16(false, a.v, false, b[t].v,
                                                         (short)0, acc[t], false, false);

    #pragma unroll
    for (int t = 0; t < 4; ++t) {
        const int nn = n0 + t * 16 + l15;
        const float bv = bias ? bias[nn] : 0.0f;
        #pragma unroll
        for (int r = 0; r < 8; ++r)
            C[(size_t)(m0 + hi * 8 + r) * N + nn] = acc[t][r] + bv;
    }
}

// ---------------- LayerNorm(q,k over full D) + head-split pack; V transposed ----------------
// grid = B*S blocks of 256; Q/K -> [b,h,s,dh] bf16, V -> Vt[b,h,dh,s] bf16
__global__ __launch_bounds__(256) void ln_pack_kernel(const float* __restrict__ qkv,
                                                      const float* __restrict__ qs,
                                                      const float* __restrict__ ks,
                                                      unsigned short* __restrict__ Qb,
                                                      unsigned short* __restrict__ Kb,
                                                      unsigned short* __restrict__ Vt) {
    const int row = blockIdx.x;            // b*S + s
    const int b   = row >> 11;
    const int s   = row & (SS - 1);
    const int tid = threadIdx.x;
    __shared__ float sh[2][8];
    const float* base = qkv + (size_t)row * F3D;

    for (int part = 0; part < 2; ++part) {
        const float* p = base + part * DD;
        float vals[8], sum = 0.f, sq = 0.f;
        #pragma unroll
        for (int i = 0; i < 8; ++i) {
            float v = p[tid + i * 256];
            vals[i] = v; sum += v; sq += v * v;
        }
        #pragma unroll
        for (int m = 16; m >= 1; m >>= 1) {
            sum += __shfl_xor(sum, m, 32);
            sq  += __shfl_xor(sq,  m, 32);
        }
        if ((tid & 31) == 0) { sh[0][tid >> 5] = sum; sh[1][tid >> 5] = sq; }
        __syncthreads();
        float ts = 0.f, tq = 0.f;
        #pragma unroll
        for (int w = 0; w < 8; ++w) { ts += sh[0][w]; tq += sh[1][w]; }
        const float mean = ts * (1.0f / DD);
        const float var  = tq * (1.0f / DD) - mean * mean;
        const float inv  = rsqrtf(var + 1e-6f);
        const float* scale = part ? ks : qs;
        unsigned short* dst = part ? Kb : Qb;
        #pragma unroll
        for (int i = 0; i < 8; ++i) {
            const int idx = tid + i * 256;
            const int hh = idx >> 7, dh = idx & 127;
            dst[((size_t)(b * HH + hh) * SS + s) * DHD + dh] = f2bf((vals[i] - mean) * inv * scale[idx]);
        }
        __syncthreads();
    }
    const float* p = base + 2 * DD;
    #pragma unroll
    for (int i = 0; i < 8; ++i) {
        const int idx = tid + i * 256;
        const int hh = idx >> 7, dh = idx & 127;
        Vt[((size_t)(b * HH + hh) * DHD + dh) * SS + s] = f2bf(p[idx]);
    }
}

// ---------------- flash attention: one wave per (b, h, 16-query tile) ----------------
__global__ __launch_bounds__(32) void attn_kernel(const unsigned short* __restrict__ Qb,
                                                  const unsigned short* __restrict__ Kb,
                                                  const unsigned short* __restrict__ Vt,
                                                  const float* __restrict__ bias,     // [H,S]
                                                  const int*   __restrict__ amask,    // [B,S]
                                                  unsigned short* __restrict__ Ob) {  // [b,s,h,dh]
    const int lane = threadIdx.x & 31;
    const int q0   = blockIdx.x * 16;
    const int h    = blockIdx.y;
    const int b    = blockIdx.z;
    const int l15  = lane & 15;
    const int hi   = lane >> 4;
    const int kbA  = hi ? 8 : 0;
    const int ksB  = hi ? 8 : 0;
    const size_t bh = (size_t)(b * HH + h);
    const float sm_scale = 0.08838834764831845f;   // 1/sqrt(128)

    // Q fragments: 4 chunks of (16 x 32) along head dim
    Frag qf[4];
    const unsigned short* qrow = Qb + (bh * SS + q0 + l15) * DHD;
    #pragma unroll
    for (int f = 0; f < 4; ++f)
        load_frag(qf[f], qrow + f * 32 + kbA, qrow + f * 32 + 16 + kbA);

    f32x8 o[8] = {};
    float rmax[8], rsum[8];
    #pragma unroll
    for (int r = 0; r < 8; ++r) { rmax[r] = -1e30f; rsum[r] = 0.f; }

    __shared__ __align__(16) unsigned short pl[16 * 32];   // P tile, bf16

    const int nkt = (q0 + 16 + 31) / 32;                   // causal: 32-key steps
    for (int j = 0; j < nkt; ++j) {
        const int kb0 = j * 32;
        // prefetch next key/value tiles (global_prefetch_b8)
        if (j + 1 < nkt) {
            __builtin_prefetch(Kb + (bh * SS + kb0 + 32 + l15) * DHD + ksB, 0, 1);
            __builtin_prefetch(Vt + (bh * DHD + l15) * SS + kb0 + 32 + ksB, 0, 1);
        }
        f32x8 sc[2] = {};
        #pragma unroll
        for (int tt = 0; tt < 2; ++tt) {
            const unsigned short* krow = Kb + (bh * SS + kb0 + tt * 16 + l15) * DHD;
            #pragma unroll
            for (int f = 0; f < 4; ++f) {
                Frag kf;
                load_frag(kf, krow + f * 32 + ksB, krow + f * 32 + ksB + 16);
                sc[tt] = __builtin_amdgcn_wmma_f32_16x16x32_bf16(false, qf[f].v, false, kf.v,
                                                                 (short)0, sc[tt], false, false);
            }
        }
        // scale + ALiBi bias + padding mask + causal mask (lane owns one key column per tile)
        #pragma unroll
        for (int tt = 0; tt < 2; ++tt) {
            const int kn = kb0 + tt * 16 + l15;
            const float badd = bias[h * SS + kn] + (amask[b * SS + kn] ? 0.f : -1e30f);
            #pragma unroll
            for (int r = 0; r < 8; ++r) {
                const int qi = q0 + hi * 8 + r;
                float v = sc[tt][r] * sm_scale + badd;
                if (kn > qi) v = -1e30f;
                sc[tt][r] = v;
            }
        }
        // online softmax (row stats replicated across each 16-lane half)
        #pragma unroll
        for (int r = 0; r < 8; ++r) {
            float mx = fmaxf(sc[0][r], sc[1][r]);
            #pragma unroll
            for (int m = 8; m >= 1; m >>= 1) mx = fmaxf(mx, __shfl_xor(mx, m, 32));
            const float mnew = fmaxf(rmax[r], mx);
            const float corr = __expf(rmax[r] - mnew);
            const float p0 = __expf(sc[0][r] - mnew);
            const float p1 = __expf(sc[1][r] - mnew);
            float ps = p0 + p1;
            #pragma unroll
            for (int m = 8; m >= 1; m >>= 1) ps += __shfl_xor(ps, m, 32);
            rsum[r] = rsum[r] * corr + ps;
            rmax[r] = mnew;
            sc[0][r] = p0; sc[1][r] = p1;
            #pragma unroll
            for (int f = 0; f < 8; ++f) o[f][r] *= corr;
        }
        // re-layout P: D-frag -> LDS -> A-frag (16 x 32)
        #pragma unroll
        for (int r = 0; r < 8; ++r) {
            const int m = hi * 8 + r;
            pl[m * 32 + l15]      = f2bf(sc[0][r]);
            pl[m * 32 + 16 + l15] = f2bf(sc[1][r]);
        }
        __syncthreads();   // single-wave WG: barrier is a HW NOP, but orders LDS
        Frag pf;
        load_frag(pf, &pl[l15 * 32 + kbA], &pl[l15 * 32 + 16 + kbA]);
        // PV: B-frag column = head-dim index -> contiguous along keys in Vt
        #pragma unroll
        for (int f = 0; f < 8; ++f) {
            const unsigned short* vrow = Vt + (bh * DHD + f * 16 + l15) * SS + kb0 + ksB;
            Frag vf;
            load_frag(vf, vrow, vrow + 16);
            o[f] = __builtin_amdgcn_wmma_f32_16x16x32_bf16(false, pf.v, false, vf.v,
                                                           (short)0, o[f], false, false);
        }
        __syncthreads();
    }
    // write O / l  -> attnout [b, s, h, dh] (bf16, A-matrix of out-proj)
    #pragma unroll
    for (int f = 0; f < 8; ++f) {
        const int nd = f * 16 + l15;
        #pragma unroll
        for (int r = 0; r < 8; ++r) {
            const int m = q0 + hi * 8 + r;
            Ob[((size_t)(b * SS + m) * HH + h) * DHD + nd] = f2bf(o[f][r] / rsum[r]);
        }
    }
}

// ---------------- launch ----------------
static inline size_t align256(size_t x) { return (x + 255) & ~(size_t)255; }

extern "C" void kernel_launch(void* const* d_in, const int* in_sizes, int n_in,
                              void* d_out, int out_size, void* d_ws, size_t ws_size,
                              hipStream_t stream) {
    if (n_in < 9) return;
    const float* x      = (const float*)d_in[0];   // [B,S,D]
    const float* abias  = (const float*)d_in[1];   // [1,H,1,S]
    const int*   amask  = (const int*)  d_in[2];   // [B,S]
    const float* w_qkv  = (const float*)d_in[3];   // [D,3D]
    const float* b_qkv  = (const float*)d_in[4];   // [3D]
    const float* wo     = (const float*)d_in[5];   // [D,D]
    const float* b_wo   = (const float*)d_in[6];   // [D]
    const float* q_ls   = (const float*)d_in[7];   // [D]
    const float* k_ls   = (const float*)d_in[8];   // [D]
    float* out          = (float*)d_out;           // [B,S,D]

    // workspace carve-up
    char* p = (char*)d_ws;
    size_t off = 0;
    auto take = [&](size_t bytes) { char* q = p + off; off += align256(bytes); return q; };
    unsigned short* xb     = (unsigned short*)take((size_t)MROWS * DD * 2);
    unsigned short* wqkvT  = (unsigned short*)take((size_t)F3D * DD * 2);
    unsigned short* woT    = (unsigned short*)take((size_t)DD * DD * 2);
    float*          qkv    = (float*)        take((size_t)MROWS * F3D * 4);
    unsigned short* Qb     = (unsigned short*)take((size_t)BB * HH * SS * DHD * 2);
    unsigned short* Kb     = (unsigned short*)take((size_t)BB * HH * SS * DHD * 2);
    unsigned short* Vt     = (unsigned short*)take((size_t)BB * HH * DHD * SS * 2);
    unsigned short* attnO  = (unsigned short*)take((size_t)MROWS * DD * 2);
    if (off > ws_size) return;   // workspace too small: nothing safe to do

    // 1) x -> bf16
    cvt_bf16_kernel<<<4096, 256, 0, stream>>>(x, xb, MROWS * DD);
    // 2) weight transposes (+bf16)
    transpose_cvt_kernel<<<dim3(F3D / 32, DD / 32), 256, 0, stream>>>(w_qkv, wqkvT, DD, F3D);
    transpose_cvt_kernel<<<dim3(DD / 32, DD / 32), 256, 0, stream>>>(wo, woT, DD, DD);
    // 3) QKV = x @ Wqkv + b   (f32 out for LN stats)
    gemm_bf16_kernel<<<dim3(F3D / 64, MROWS / 128), 256, 0, stream>>>(xb, wqkvT, b_qkv, qkv,
                                                                      MROWS, F3D, DD);
    // 4) LN(q), LN(k), head-split; V transposed
    ln_pack_kernel<<<MROWS, 256, 0, stream>>>(qkv, q_ls, k_ls, Qb, Kb, Vt);
    // 5) flash attention
    attn_kernel<<<dim3(SS / 16, HH, BB), 32, 0, stream>>>(Qb, Kb, Vt, abias, amask, attnO);
    // 6) out = attnO @ Wo + b_wo
    gemm_bf16_kernel<<<dim3(DD / 64, MROWS / 128), 256, 0, stream>>>(attnO, woT, b_wo, out,
                                                                     MROWS, DD, DD);
}
